// EnhancedLSTM_81277961109817
// MI455X (gfx1250) — matrix operand order repaired and checked by reading, not compiled, not run on hardware
//
#include <hip/hip_runtime.h>

// ---------------------------------------------------------------------------
// CDNA5 (gfx1250, wave32) implementation of the xLSTM reference.
// All matmuls use v_wmma_f32_16x16x32_bf16 (bf16 A/B, f32 accumulate).
// Fragments load as contiguous 16B/32B chunks (ds_load_b128); global->LDS
// staging is register-prefetched across iterations to overlap with WMMAs.
// GEMM: 128x64 workgroup tile, 4 waves, 8 WMMAs per wave per k-step.
// ---------------------------------------------------------------------------

#define Bb   16
#define Ss   512
#define Dd   512
#define Ll   4
#define Hh   8
#define Oo   1
#define QKd  256          // D/2
#define Vv   512
#define Ff   1408
#define DQK  32
#define DV   64
#define NTOK (Bb*Ss)      // 8192
#define EPSf 1e-6f
#define CAPf 15.0f
#define BN_EPSf 1e-5f

typedef __bf16 bf16;
typedef __attribute__((ext_vector_type(16))) __bf16 v16bf;
typedef __attribute__((ext_vector_type(8)))  float  v8f;
typedef __attribute__((ext_vector_type(4)))  unsigned int v4u;   // 16 bytes

__device__ __forceinline__ v8f wmma_bf16(v16bf a, v16bf b, v8f c) {
  // D = A(16x32) * B(32x16) + C(16x16), f32 accumulate
  return __builtin_amdgcn_wmma_f32_16x16x32_bf16(false, a, false, b, (short)0, c,
                                                 false, false);
}

// A-fragment (contiguous layout): elements = p[2*kb..2*kb+15]
// -> two 16-byte loads.
__device__ __forceinline__ v16bf load_a_frag(const bf16* __restrict__ rowp, int kb) {
  union { v4u u[2]; v16bf v; } r;
  r.u[0] = *(const v4u*)(rowp + 2 * kb);
  r.u[1] = *(const v4u*)(rowp + 2 * kb + 8);
  return r.v;
}

// B-fragment: 16 contiguous bf16 (32 bytes, 32B-aligned) -> two ds_load_b128.
__device__ __forceinline__ v16bf load_b_frag(const bf16* __restrict__ p) {
  return *(const v16bf*)p;
}

// ---------------------------------------------------------------------------
// Tiled transpose + fp32->bf16 convert:  in (l, K, N) f32  ->  out (l, N, K) bf16
// grid = (N/16, K/16, L), block = 256
// ---------------------------------------------------------------------------
__global__ __launch_bounds__(256) void cvt_t_kernel(const float* __restrict__ in,
                                                    bf16* __restrict__ out,
                                                    int K, int N) {
  __shared__ float t[16][17];
  int l = blockIdx.z;
  const float* ip = in + (size_t)l * K * N;
  bf16* op = out + (size_t)l * K * N;
  int n0 = blockIdx.x * 16, k0 = blockIdx.y * 16;
  int tx = threadIdx.x & 15, ty = threadIdx.x >> 4;
  t[ty][tx] = ip[(size_t)(k0 + ty) * N + n0 + tx];
  __syncthreads();
  op[(size_t)(n0 + ty) * K + k0 + tx] = (bf16)t[tx][ty];
}

// ---------------------------------------------------------------------------
// Input batch-norm over (B,S) per channel; writes xn and h (= xn)
// ---------------------------------------------------------------------------
__global__ __launch_bounds__(256) void bn_in_kernel(
    const float* __restrict__ x, const float* __restrict__ gamma,
    const float* __restrict__ beta, float* __restrict__ xn, float* __restrict__ h) {
  int c = blockIdx.x, tid = threadIdx.x;
  float s = 0.f, s2 = 0.f;
  for (int r = tid; r < NTOK; r += 256) {
    float v = x[(size_t)r * Dd + c];
    s += v; s2 += v * v;
  }
  __shared__ float rs[8], rs2[8], stats[2];
  for (int m = 1; m < 32; m <<= 1) { s += __shfl_xor(s, m, 32); s2 += __shfl_xor(s2, m, 32); }
  if ((tid & 31) == 0) { rs[tid >> 5] = s; rs2[tid >> 5] = s2; }
  __syncthreads();
  if (tid == 0) {
    float a = 0.f, b2 = 0.f;
    for (int i = 0; i < 8; ++i) { a += rs[i]; b2 += rs2[i]; }
    float mu = a / (float)NTOK;
    float var = b2 / (float)NTOK - mu * mu;
    stats[0] = mu; stats[1] = rsqrtf(var + BN_EPSf);
  }
  __syncthreads();
  float mu = stats[0], iv = stats[1] * gamma[c], be = beta[c];
  for (int r = tid; r < NTOK; r += 256) {
    float v = (x[(size_t)r * Dd + c] - mu) * iv + be;
    xn[(size_t)r * Dd + c] = v;
    h[(size_t)r * Dd + c] = v;
  }
}

// ---------------------------------------------------------------------------
// RMS-norm: row (base + blockIdx*stride) of x -> out_bf / out_f (indexed by blockIdx)
// ---------------------------------------------------------------------------
__global__ __launch_bounds__(256) void rmsnorm_kernel(
    const float* __restrict__ x, const float* __restrict__ w,
    bf16* __restrict__ out_bf, float* __restrict__ out_f, int base, int stride) {
  int rowIn = base + blockIdx.x * stride;
  const float* xr = x + (size_t)rowIn * Dd;
  int tid = threadIdx.x;
  float ss = 0.f;
  for (int d = tid; d < Dd; d += 256) { float v = xr[d]; ss += v * v; }
  __shared__ float red[8];
  for (int m = 1; m < 32; m <<= 1) ss += __shfl_xor(ss, m, 32);
  if ((tid & 31) == 0) red[tid >> 5] = ss;
  __syncthreads();
  if (tid == 0) {
    float t = 0.f;
    for (int i = 0; i < 8; ++i) t += red[i];
    red[0] = rsqrtf(t / (float)Dd + EPSf);
  }
  __syncthreads();
  float r = red[0];
  for (int d = tid; d < Dd; d += 256) {
    float v = xr[d] * r * w[d];
    if (out_bf) out_bf[(size_t)blockIdx.x * Dd + d] = (bf16)v;
    if (out_f)  out_f [(size_t)blockIdx.x * Dd + d] = v;
  }
}

// ---------------------------------------------------------------------------
// Generic bf16 WMMA GEMM: C[M,N] (f32) = A[M,K] @ Bt[N,K]^T   (Bt pre-transposed)
// block = 128 threads = 4 waves; workgroup tile 128x64.
// Each wave owns two 16-row stripes (rows wv*32 .. wv*32+31): 8 WMMA / k-step.
// Register-prefetch of the next k-tile overlaps HBM latency with WMMAs.
// mode 0: C = acc        mode 1: C += acc
// mode 2: Obf = acc      mode 3: Obf = silu(acc) * aux    (gated FFN epilogue)
// ---------------------------------------------------------------------------
__global__ __launch_bounds__(128) void gemm_kernel(
    const bf16* __restrict__ A, const bf16* __restrict__ Bt,
    float* __restrict__ C, bf16* __restrict__ Obf, const float* __restrict__ aux,
    int M, int N, int K, int mode) {
  __shared__ v4u As4[512];   // 128 rows x 32 k  (bf16)
  __shared__ v4u Bs4[256];   //  64 n-rows x 32 k (bf16, B already transposed)
  const bf16* Asb = (const bf16*)As4;
  const bf16* Bsb = (const bf16*)Bs4;
  int m0 = blockIdx.x * 128, n0 = blockIdx.y * 64;
  int tid = threadIdx.x, lane = tid & 31, wv = tid >> 5;
  int col = lane & 15;
  int kb   = (lane < 16) ? 0 : 8;    // A-frag K base / 2 (contiguous: k = 2*kb + s)
  int kh16 = (lane < 16) ? 0 : 16;   // B-frag K-half base
  // per-thread staging coordinates (fully static: no exec-mask branching)
  int rs = tid >> 2, cs = (tid & 3) * 8;
  const bf16* Ap = A  + (size_t)(m0 + rs) * K + cs;   // rows rs, rs+32, rs+64, rs+96
  const bf16* Bp = Bt + (size_t)(n0 + rs) * K + cs;   // rows rs, rs+32
  v8f acc[2][4] = {{v8f{}, v8f{}, v8f{}, v8f{}}, {v8f{}, v8f{}, v8f{}, v8f{}}};
  v4u ra0, ra1, ra2, ra3, rb0, rb1;
  // prologue fetch (k0 = 0)
  ra0 = *(const v4u*)(Ap);
  ra1 = *(const v4u*)(Ap + (size_t)32 * K);
  ra2 = *(const v4u*)(Ap + (size_t)64 * K);
  ra3 = *(const v4u*)(Ap + (size_t)96 * K);
  rb0 = *(const v4u*)(Bp);
  rb1 = *(const v4u*)(Bp + (size_t)32 * K);
  for (int k0 = 0; k0 < K; k0 += 32) {
    As4[tid]       = ra0;
    As4[tid + 128] = ra1;
    As4[tid + 256] = ra2;
    As4[tid + 384] = ra3;
    Bs4[tid]       = rb0;
    Bs4[tid + 128] = rb1;
    __syncthreads();
    if (k0 + 32 < K) {       // prefetch next tile; overlaps with the 8 WMMAs
      const bf16* An = Ap + k0 + 32;
      const bf16* Bn = Bp + k0 + 32;
      ra0 = *(const v4u*)(An);
      ra1 = *(const v4u*)(An + (size_t)32 * K);
      ra2 = *(const v4u*)(An + (size_t)64 * K);
      ra3 = *(const v4u*)(An + (size_t)96 * K);
      rb0 = *(const v4u*)(Bn);
      rb1 = *(const v4u*)(Bn + (size_t)32 * K);
    }
    v16bf af0 = load_a_frag(Asb + (wv * 32 + col) * 32, kb);
    v16bf af1 = load_a_frag(Asb + (wv * 32 + 16 + col) * 32, kb);
#pragma unroll
    for (int t = 0; t < 4; ++t) {
      v16bf bfm = load_b_frag(Bsb + (t * 16 + col) * 32 + kh16);
      acc[0][t] = wmma_bf16(af0, bfm, acc[0][t]);
      acc[1][t] = wmma_bf16(af1, bfm, acc[1][t]);
    }
    __syncthreads();
  }
#pragma unroll
  for (int s = 0; s < 2; ++s) {
    int rbase = m0 + wv * 32 + s * 16 + ((lane < 16) ? 0 : 8);
#pragma unroll
    for (int t = 0; t < 4; ++t) {
      int n = n0 + t * 16 + col;
#pragma unroll
      for (int i = 0; i < 8; ++i) {
        size_t idx = (size_t)(rbase + i) * N + n;
        float v = acc[s][t][i];
        if (mode == 0)      C[idx] = v;
        else if (mode == 1) C[idx] += v;
        else if (mode == 2) Obf[idx] = (bf16)v;
        else {
          float sv = v / (1.f + __expf(-v));   // silu
          Obf[idx] = (bf16)(sv * aux[idx]);
        }
      }
    }
  }
}

// ---------------------------------------------------------------------------
// Gate projections: ig/fg = soft_cap(y @ Wi/Wf + b), stored [b*H+h][S]
// ---------------------------------------------------------------------------
__global__ __launch_bounds__(256) void gates_proj_kernel(
    const float* __restrict__ y, const float* __restrict__ Wi,
    const float* __restrict__ bi, const float* __restrict__ Wf,
    const float* __restrict__ bfv, float* __restrict__ ig, float* __restrict__ fg) {
  int unit = blockIdx.x * 8 + (threadIdx.x >> 5);  // token*H + h
  int lane = threadIdx.x & 31;
  int token = unit / Hh, h = unit % Hh;
  const float* yr = y + (size_t)token * Dd;
  float si = 0.f, sf = 0.f;
  for (int d = lane; d < Dd; d += 32) {
    float yv = yr[d];
    si += yv * Wi[(size_t)d * Hh + h];
    sf += yv * Wf[(size_t)d * Hh + h];
  }
  for (int m = 1; m < 32; m <<= 1) { si += __shfl_xor(si, m, 32); sf += __shfl_xor(sf, m, 32); }
  if (lane == 0) {
    si = CAPf * tanhf((si + bi[h]) / CAPf);
    sf = CAPf * tanhf((sf + bfv[h]) / CAPf);
    int b = token / Ss, s = token % Ss;
    size_t idx = ((size_t)b * Hh + h) * Ss + s;
    ig[idx] = si; fg[idx] = sf;
  }
}

// ---------------------------------------------------------------------------
// Per-(b,h) decay scan.  b_t = cumsum(log_sigmoid(fg)); c_s = ig_s - b_s;
// m_t = b_t + cummax(c); rho_t = exp(b_t - m_t); kappa_s = exp(c_s); enm = exp(-m)
// ---------------------------------------------------------------------------
__global__ void gates_scan_kernel(const float* __restrict__ ig,
                                  const float* __restrict__ fg,
                                  float* __restrict__ rho, float* __restrict__ enm,
                                  float* __restrict__ kap) {
  if (threadIdx.x != 0) return;
  size_t base = (size_t)blockIdx.x * Ss;
  float bcum = 0.f, cmax = -3.0e38f;
  for (int s = 0; s < Ss; ++s) {
    float f = fg[base + s];
    float lsig = fminf(f, 0.f) - log1pf(__expf(-fabsf(f)));  // log_sigmoid
    bcum += lsig;
    float c = ig[base + s] - bcum;
    cmax = fmaxf(cmax, c);
    float m = bcum + cmax;
    rho[base + s] = __expf(bcum - m);
    enm[base + s] = __expf(-m);
    kap[base + s] = __expf(c);
  }
}

// ---------------------------------------------------------------------------
// Pack q/k/v to per-(b,h) bf16; fold kappa & DQK^-0.5 into k; v stored transposed
// vh layout: [b*H+h][d][s]   (so attention B-fragments are contiguous)
// ---------------------------------------------------------------------------
__global__ void pack_qkv_kernel(const float* __restrict__ q, const float* __restrict__ k,
                                const float* __restrict__ v, const float* __restrict__ kap,
                                bf16* __restrict__ qh, bf16* __restrict__ kh,
                                bf16* __restrict__ vh) {
  int idx = blockIdx.x * 256 + threadIdx.x;           // over NTOK*H*DV
  if (idx >= NTOK * Hh * DV) return;
  int d = idx & 63;
  int rem = idx >> 6;
  int h = rem % Hh, token = rem / Hh;
  int b = token / Ss, s = token % Ss;
  size_t bh = (size_t)b * Hh + h;
  vh[(bh * DV + d) * Ss + s] = (bf16)v[(size_t)token * Vv + h * DV + d];
  if (d < DQK) {
    size_t dsti = bh * Ss + s;
    float kscale = kap[dsti] * 0.17677669529663687f;  // kappa * DQK^-0.5
    qh[dsti * DQK + d] = (bf16)q[(size_t)token * QKd + h * DQK + d];
    kh[dsti * DQK + d] = (bf16)(k[(size_t)token * QKd + h * DQK + d] * kscale);
  }
}

// ---------------------------------------------------------------------------
// Fused mLSTM attention.  grid = (S/64, B*H), block = 128 (4 waves).
// Each wave owns 16 query rows.  scores = (q@k'^T) * rho_t, causal-masked;
// norm = max(|rowsum|, exp(-m)); ht = (scores @ v) / (norm + eps).
// 12 WMMAs per wave per key tile; kT/vT staging register-prefetched.
// ---------------------------------------------------------------------------
__global__ __launch_bounds__(128) void attn_kernel(
    const bf16* __restrict__ qh, const bf16* __restrict__ kh,
    const bf16* __restrict__ vh, const float* __restrict__ rho,
    const float* __restrict__ enm, float* __restrict__ ht) {
  int bh = blockIdx.y;
  int tq = blockIdx.x;
  int t0 = tq * 64;
  __shared__ v4u kT4[256];                 // 64 s-rows x 32 k (= k^T tile)
  __shared__ v4u vT4[512];                 // 64 d-rows x 64 s (v^T tile)
  __shared__ alignas(16) bf16 sStage[4][16 * 64];
  __shared__ float rhoS[64], enmS[64];
  const bf16* kTb = (const bf16*)kT4;
  const bf16* vTb = (const bf16*)vT4;
  int tid = threadIdx.x, lane = tid & 31, wv = tid >> 5;
  int col = lane & 15;
  int kb   = (lane < 16) ? 0 : 8;
  int kh16 = (lane < 16) ? 0 : 16;
  int roff = (lane < 16) ? 0 : 8;
  if (tid < 64) {
    rhoS[tid] = rho[(size_t)bh * Ss + t0 + tid];
    enmS[tid] = enm[(size_t)bh * Ss + t0 + tid];
  }
  // q A-fragment (persistent, direct from global)
  v16bf aq = load_a_frag(qh + ((size_t)bh * Ss + t0 + wv * 16 + col) * DQK, kb);
  v8f out[4] = {v8f{}, v8f{}, v8f{}, v8f{}};
  float rowsum[8] = {0, 0, 0, 0, 0, 0, 0, 0};
  // per-thread staging coordinates
  int rk = tid >> 2, ck = (tid & 3) * 8;            // kT: rows rk, rk+32
  int rv = tid >> 3, cv = (tid & 7) * 8;            // vT: d-rows rv, rv+16, rv+32, rv+48
  const bf16* kBase = kh + ((size_t)bh * Ss + rk) * DQK + ck;
  const bf16* vBase = vh + ((size_t)bh * DV + rv) * Ss + cv;
  v4u fk0, fk1, fv0, fv1, fv2, fv3;
  // prologue fetch (jt = 0)
  fk0 = *(const v4u*)(kBase);
  fk1 = *(const v4u*)(kBase + (size_t)32 * DQK);
  fv0 = *(const v4u*)(vBase);
  fv1 = *(const v4u*)(vBase + (size_t)16 * Ss);
  fv2 = *(const v4u*)(vBase + (size_t)32 * Ss);
  fv3 = *(const v4u*)(vBase + (size_t)48 * Ss);
  for (int jt = 0; jt <= tq; ++jt) {
    int s0 = jt * 64;
    kT4[tid]       = fk0;
    kT4[tid + 128] = fk1;
    vT4[tid]       = fv0;
    vT4[tid + 128] = fv1;
    vT4[tid + 256] = fv2;
    vT4[tid + 384] = fv3;
    __syncthreads();
    if (jt < tq) {            // prefetch next key/value tile
      const bf16* kn = kBase + (size_t)(s0 + 64) * DQK;
      const bf16* vn = vBase + s0 + 64;
      fk0 = *(const v4u*)(kn);
      fk1 = *(const v4u*)(kn + (size_t)32 * DQK);
      fv0 = *(const v4u*)(vn);
      fv1 = *(const v4u*)(vn + (size_t)16 * Ss);
      fv2 = *(const v4u*)(vn + (size_t)32 * Ss);
      fv3 = *(const v4u*)(vn + (size_t)48 * Ss);
    }
    // --- scores = q @ k'^T, scale/mask epilogue, stage as bf16 ---
#pragma unroll
    for (int t = 0; t < 4; ++t) {
      v16bf bk = load_b_frag(kTb + (t * 16 + col) * 32 + kh16);
      v8f sf = wmma_bf16(aq, bk, v8f{});
      bf16* stg = &sStage[wv][0];
#pragma unroll
      for (int i = 0; i < 8; ++i) {
        int rloc = wv * 16 + i + roff;
        int tg = t0 + rloc;
        int sg = s0 + t * 16 + col;
        float v = sf[i] * rhoS[rloc];
        if (sg > tg) v = 0.f;
        rowsum[i] += v;
        stg[(i + roff) * 64 + t * 16 + col] = (bf16)v;
      }
    }
    // --- out += scores @ v ---
    const bf16* sp = &sStage[wv][col * 64];
    v16bf a0 = load_a_frag(sp, kb);
    v16bf a1 = load_a_frag(sp + 32, kb);
#pragma unroll
    for (int n = 0; n < 4; ++n) {
      v16bf b0 = load_b_frag(vTb + (n * 16 + col) * 64 + kh16);
      v16bf b1 = load_b_frag(vTb + (n * 16 + col) * 64 + 32 + kh16);
      out[n] = wmma_bf16(a0, b0, out[n]);
      out[n] = wmma_bf16(a1, b1, out[n]);
    }
    __syncthreads();
  }
  // reduce signed row-sums across the 16 column lanes of each half-wave
#pragma unroll
  for (int i = 0; i < 8; ++i) {
    float r = rowsum[i];
    for (int m = 1; m < 16; m <<= 1) r += __shfl_xor(r, m, 32);
    rowsum[i] = r;
  }
  int b = bh / Hh, h = bh % Hh;
#pragma unroll
  for (int i = 0; i < 8; ++i) {
    int tl = wv * 16 + i + roff;
    float nrm = fmaxf(fabsf(rowsum[i]), enmS[tl]);
    float sc = 1.f / (nrm + EPSf);
    size_t rowo = (((size_t)b * Ss + (t0 + tl)) * Hh + h) * DV;
#pragma unroll
    for (int n = 0; n < 4; ++n) ht[rowo + n * 16 + col] = out[n][i] * sc;
  }
}

// ---------------------------------------------------------------------------
// Per-(token, head) layer-norm of ht, * mhn_w, * sigmoid(og) -> act (bf16)
// ---------------------------------------------------------------------------
__global__ __launch_bounds__(64) void mhn_kernel(
    const float* __restrict__ ht, const float* __restrict__ ograw,
    const float* __restrict__ mhnw, bf16* __restrict__ act) {
  int unit = blockIdx.x;        // token*H + h
  int h = unit % Hh, token = unit / Hh;
  int d = threadIdx.x;
  __shared__ float part[2][2];
  float v = ht[(size_t)unit * DV + d];
  float s = v;
  for (int m = 1; m < 32; m <<= 1) s += __shfl_xor(s, m, 32);
  if ((d & 31) == 0) part[0][d >> 5] = s;
  __syncthreads();
  float mu = (part[0][0] + part[0][1]) * (1.f / 64.f);
  float dv = v - mu;
  float q = dv * dv;
  for (int m = 1; m < 32; m <<= 1) q += __shfl_xor(q, m, 32);
  if ((d & 31) == 0) part[1][d >> 5] = q;
  __syncthreads();
  float var = (part[1][0] + part[1][1]) * (1.f / 64.f);
  float hn = dv * rsqrtf(var + EPSf) * mhnw[h * DV + d];
  float og = ograw[(size_t)token * Vv + h * DV + d];
  act[(size_t)token * Vv + h * DV + d] = (bf16)(hn / (1.f + __expf(-og)));
}

// ---------------------------------------------------------------------------
// Final head: bn1(16 rows) -> fc1 -> bn2 -> gelu -> fc2.  Single workgroup.
// ---------------------------------------------------------------------------
__global__ __launch_bounds__(256) void head_kernel(
    const float* __restrict__ lastln, const float* __restrict__ bn1g,
    const float* __restrict__ bn1b, const float* __restrict__ fc1W,
    const float* __restrict__ fc1b, const float* __restrict__ bn2g,
    const float* __restrict__ bn2b, const float* __restrict__ fc2W,
    const float* __restrict__ fc2b, float* __restrict__ obuf) {
  __shared__ float a[16 * 512];
  __shared__ float o1[16 * 256];
  int tid = threadIdx.x;
  for (int c = tid; c < 512; c += 256) {
    float s = 0.f, s2 = 0.f;
    for (int b = 0; b < 16; ++b) { float v = lastln[b * 512 + c]; s += v; s2 += v * v; }
    float mu = s * (1.f / 16.f);
    float var = s2 * (1.f / 16.f) - mu * mu;
    float iv = rsqrtf(var + BN_EPSf) * bn1g[c];
    for (int b = 0; b < 16; ++b) a[b * 512 + c] = (lastln[b * 512 + c] - mu) * iv + bn1b[c];
  }
  __syncthreads();
  for (int j = tid; j < 256; j += 256) {
    float colv[16];
    for (int b = 0; b < 16; ++b) {
      float s = fc1b[j];
      const float* ar = &a[b * 512];
      for (int d = 0; d < 512; ++d) s += ar[d] * fc1W[d * 256 + j];
      colv[b] = s;
    }
    float s = 0.f, s2 = 0.f;
    for (int b = 0; b < 16; ++b) { s += colv[b]; s2 += colv[b] * colv[b]; }
    float mu = s * (1.f / 16.f);
    float var = s2 * (1.f / 16.f) - mu * mu;
    float iv = rsqrtf(var + BN_EPSf) * bn2g[j];
    for (int b = 0; b < 16; ++b) {
      float v = (colv[b] - mu) * iv + bn2b[j];
      v = 0.5f * v * (1.f + erff(v * 0.7071067811865475f));  // exact gelu
      o1[b * 256 + j] = v;
    }
  }
  __syncthreads();
  if (tid < 16) {
    float s = fc2b[0];
    for (int j = 0; j < 256; ++j) s += o1[tid * 256 + j] * fc2W[j];
    obuf[tid] = s;
  }
}

// ---------------------------------------------------------------------------
// out[b,s] = relu(o[b] + xn[b,s,:] . res_W + res_b).  One wave per row.
// ---------------------------------------------------------------------------
__global__ __launch_bounds__(256) void res_out_kernel(
    const float* __restrict__ xn, const float* __restrict__ resW,
    const float* __restrict__ resb, const float* __restrict__ obuf,
    float* __restrict__ out) {
  int row = blockIdx.x * 8 + (threadIdx.x >> 5);
  int lane = threadIdx.x & 31;
  const float* xr = xn + (size_t)row * Dd;
  float s = 0.f;
  for (int d = lane; d < Dd; d += 32) s += xr[d] * resW[d];
  for (int m = 1; m < 32; m <<= 1) s += __shfl_xor(s, m, 32);
  if (lane == 0) {
    int b = row / Ss;
    float v = obuf[b] + s + resb[0];
    out[row] = fmaxf(v, 0.f);
  }
}

// ---------------------------------------------------------------------------
// Host orchestration
// ---------------------------------------------------------------------------
extern "C" void kernel_launch(void* const* d_in, const int* in_sizes, int n_in,
                              void* d_out, int out_size, void* d_ws, size_t ws_size,
                              hipStream_t stream) {
  (void)in_sizes; (void)n_in; (void)out_size; (void)ws_size;
  const float* x        = (const float*)d_in[0];
  const float* bn_in_g  = (const float*)d_in[1];
  const float* bn_in_b  = (const float*)d_in[2];
  const float* norm1_w  = (const float*)d_in[3];
  const float* norm2_w  = (const float*)d_in[4];
  const float* Wq       = (const float*)d_in[5];
  const float* Wk       = (const float*)d_in[6];
  const float* Wv       = (const float*)d_in[7];
  const float* Wog      = (const float*)d_in[8];
  const float* Wi       = (const float*)d_in[9];
  const float* bi       = (const float*)d_in[10];
  const float* Wf       = (const float*)d_in[11];
  const float* bfv      = (const float*)d_in[12];
  const float* mhn_w    = (const float*)d_in[13];
  const float* Wout     = (const float*)d_in[14];
  const float* Wg       = (const float*)d_in[15];
  const float* Wu       = (const float*)d_in[16];
  const float* Wd       = (const float*)d_in[17];
  const float* fnorm_w  = (const float*)d_in[18];
  const float* bn1_g    = (const float*)d_in[19];
  const float* bn1_b    = (const float*)d_in[20];
  const float* fc1_W    = (const float*)d_in[21];
  const float* fc1_b    = (const float*)d_in[22];
  const float* bn2_g    = (const float*)d_in[23];
  const float* bn2_b    = (const float*)d_in[24];
  const float* fc2_W    = (const float*)d_in[25];
  const float* fc2_b    = (const float*)d_in[26];
  const float* res_W    = (const float*)d_in[27];
  const float* res_b    = (const float*)d_in[28];

  // workspace bump allocator
  char* p = (char*)d_ws;
  auto alloc = [&](size_t bytes) -> void* {
    void* r = (void*)p;
    p += (bytes + 255) & ~(size_t)255;
    return r;
  };
  const size_t BHS = (size_t)Bb * Hh * Ss;
  float* xn   = (float*)alloc((size_t)NTOK * Dd * 4);
  float* hbuf = (float*)alloc((size_t)NTOK * Dd * 4);
  float* bufC = (float*)alloc((size_t)NTOK * Dd * 4);   // y_f32, later ht
  float* qf   = (float*)alloc((size_t)NTOK * QKd * 4);
  float* kf   = (float*)alloc((size_t)NTOK * QKd * 4);
  float* vf   = (float*)alloc((size_t)NTOK * Vv * 4);
  float* ogf  = (float*)alloc((size_t)NTOK * Vv * 4);
  float* uf   = (float*)alloc((size_t)NTOK * Ff * 4);
  float* rho  = (float*)alloc(BHS * 4);
  float* enm  = (float*)alloc(BHS * 4);
  float* kap  = (float*)alloc(BHS * 4);
  float* igb  = (float*)alloc(BHS * 4);
  float* fgb  = (float*)alloc(BHS * 4);
  float* lastln = (float*)alloc(16 * 512 * 4);
  float* obuf   = (float*)alloc(64 * 4);
  bf16* ybf = (bf16*)alloc((size_t)NTOK * Dd * 2);      // y / z / act bf16
  bf16* qh  = (bf16*)alloc(BHS * DQK * 2);
  bf16* kh  = (bf16*)alloc(BHS * DQK * 2);
  bf16* vh  = (bf16*)alloc(BHS * DV * 2);               // transposed [bh][d][s]
  bf16* gu  = (bf16*)alloc((size_t)NTOK * Ff * 2);
  const size_t nWq = (size_t)Ll * Dd * QKd, nWv = (size_t)Ll * Dd * Vv,
               nWg = (size_t)Ll * Dd * Ff;
  bf16* WqB  = (bf16*)alloc(nWq * 2);    // all weight copies are TRANSPOSED (N x K)
  bf16* WkB  = (bf16*)alloc(nWq * 2);
  bf16* WvB  = (bf16*)alloc(nWv * 2);
  bf16* WogB = (bf16*)alloc(nWv * 2);
  bf16* WoutB= (bf16*)alloc(nWv * 2);
  bf16* WgB  = (bf16*)alloc(nWg * 2);
  bf16* WuB  = (bf16*)alloc(nWg * 2);
  bf16* WdB  = (bf16*)alloc(nWg * 2);

  auto cvtT = [&](const float* src, bf16* dst, int K, int N) {
    cvt_t_kernel<<<dim3(N / 16, K / 16, Ll), 256, 0, stream>>>(src, dst, K, N);
  };
  cvtT(Wq, WqB, Dd, QKd);   cvtT(Wk, WkB, Dd, QKd);
  cvtT(Wv, WvB, Dd, Vv);    cvtT(Wog, WogB, Dd, Vv);  cvtT(Wout, WoutB, Vv, Dd);
  cvtT(Wg, WgB, Dd, Ff);    cvtT(Wu, WuB, Dd, Ff);    cvtT(Wd, WdB, Ff, Dd);

  // input batchnorm -> xn, h
  bn_in_kernel<<<Dd, 256, 0, stream>>>(x, bn_in_g, bn_in_b, xn, hbuf);

  auto gemm = [&](const bf16* A, const bf16* Bt, float* C, bf16* Obf,
                  const float* aux, int M, int N, int K, int mode) {
    gemm_kernel<<<dim3(M / 128, N / 64), 128, 0, stream>>>(A, Bt, C, Obf, aux, M, N, K, mode);
  };

  for (int l = 0; l < Ll; ++l) {
    // y = rms_norm(h, norm1_w[l])
    rmsnorm_kernel<<<NTOK, 256, 0, stream>>>(hbuf, norm1_w + (size_t)l * Dd, ybf, bufC, 0, 1);
    // projections
    gemm(ybf, WqB  + (size_t)l * Dd * QKd, qf,  nullptr, nullptr, NTOK, QKd, Dd, 0);
    gemm(ybf, WkB  + (size_t)l * Dd * QKd, kf,  nullptr, nullptr, NTOK, QKd, Dd, 0);
    gemm(ybf, WvB  + (size_t)l * Dd * Vv,  vf,  nullptr, nullptr, NTOK, Vv,  Dd, 0);
    gemm(ybf, WogB + (size_t)l * Dd * Vv,  ogf, nullptr, nullptr, NTOK, Vv,  Dd, 0);
    // gates
    gates_proj_kernel<<<(NTOK * Hh) / 8, 256, 0, stream>>>(
        bufC, Wi + (size_t)l * Dd * Hh, bi + (size_t)l * Hh,
        Wf + (size_t)l * Dd * Hh, bfv + (size_t)l * Hh, igb, fgb);
    gates_scan_kernel<<<Bb * Hh, 32, 0, stream>>>(igb, fgb, rho, enm, kap);
    pack_qkv_kernel<<<(NTOK * Hh * DV) / 256, 256, 0, stream>>>(qf, kf, vf, kap, qh, kh, vh);
    // attention -> ht (bufC)
    attn_kernel<<<dim3(Ss / 64, Bb * Hh), 128, 0, stream>>>(qh, kh, vh, rho, enm, bufC);
    // multihead norm * sigmoid(og) -> act (ybf)
    mhn_kernel<<<NTOK * Hh, 64, 0, stream>>>(bufC, ogf, mhn_w + (size_t)l * Vv, ybf);
    // h += act @ Wout[l]
    gemm(ybf, WoutB + (size_t)l * Vv * Dd, hbuf, nullptr, nullptr, NTOK, Dd, Vv, 1);
    // FFN: z = rms_norm(h); h += (silu(z@Wg) * (z@Wu)) @ Wd
    rmsnorm_kernel<<<NTOK, 256, 0, stream>>>(hbuf, norm2_w + (size_t)l * Dd, ybf, nullptr, 0, 1);
    gemm(ybf, WuB + (size_t)l * Dd * Ff, uf, nullptr, nullptr, NTOK, Ff, Dd, 0);
    gemm(ybf, WgB + (size_t)l * Dd * Ff, nullptr, gu, uf, NTOK, Ff, Dd, 3);
    gemm(gu,  WdB + (size_t)l * Ff * Dd, hbuf, nullptr, nullptr, NTOK, Dd, Ff, 1);
  }

  // final norm on last-token rows only (reference only uses h[:, -1, :])
  rmsnorm_kernel<<<Bb, 256, 0, stream>>>(hbuf, fnorm_w, nullptr, lastln, Ss - 1, Ss);
  head_kernel<<<1, 256, 0, stream>>>(lastln, bn1_g, bn1_b, fc1_W, fc1_b,
                                     bn2_g, bn2_b, fc2_W, fc2_b, obuf);
  res_out_kernel<<<NTOK / 8, 256, 0, stream>>>(xn, res_W, res_b, obuf, (float*)d_out);
}
